// MiniSTU_9715216024185
// MI455X (gfx1250) — matrix operand order repaired and checked by reading, not compiled, not run on hardware
//
#include <hip/hip_runtime.h>
#include <hip/hip_bf16.h>

// ---------------------------------------------------------------------------
// MiniSTU on MI455X (gfx1250, wave32).
// Phase 1 (transposed Toeplitz conv):  C[i][l] = sum_t xT[i][t] * T[t][l],
//   T[t][l] = phi[l-t] (0 for l<t) via reversed zero-padded filter table
//   (causal mask is data, not control flow).  Filter table staged to LDS
//   with the Tensor Data Mover (tensor_load_to_lds + s_wait_tensorcnt).
// Phase 2: projection GEMM [B*L, 12288] x [12288, 256] -> out (f32),
//   U streamed from HBM exactly once.
// All matrix math: v_wmma_f32_16x16x32_bf16 (bf16 in, f32 accumulate).
// ---------------------------------------------------------------------------

typedef __attribute__((ext_vector_type(16))) __bf16       v16bf;
typedef __attribute__((ext_vector_type(8)))  float        v8f;
typedef __attribute__((ext_vector_type(4)))  unsigned int v4u;
typedef __attribute__((ext_vector_type(8)))  int          v8i;
typedef __attribute__((ext_vector_type(4)))  int          v4i;

#define SEQ   1024
#define NF    24
#define IDIM  256
#define ODIM  256
#define BATCH 4
#define KDIM  (2 * NF * IDIM)          // 12288
#define PHIN  1152                     // padded reversed filter length
#define PHID  1087                     // phiRev[i] = phi[PHID - i], zero outside

// ---- workspace layout (bytes), all 16B aligned ----
static constexpr size_t OFF_XT     = 0;                                          // [B][I][SEQ] bf16
static constexpr size_t OFF_PHIREP = OFF_XT     + (size_t)BATCH * IDIM * SEQ * 2; // [2*NF][8][PHIN] bf16
static constexpr size_t OFF_MT     = OFF_PHIREP + (size_t)2 * NF * 8 * PHIN * 2;  // [ODIM][KDIM] bf16
static constexpr size_t OFF_UCAT   = OFF_MT     + (size_t)ODIM * KDIM * 2;        // [B*L][KDIM] bf16

// ---------------------------------------------------------------------------
// prep kernels
// ---------------------------------------------------------------------------

__global__ void prep_xT_kernel(const float* __restrict__ x, __bf16* __restrict__ xT) {
    int idx = blockIdx.x * blockDim.x + threadIdx.x;     // over B*I*SEQ, t fastest
    if (idx >= BATCH * IDIM * SEQ) return;
    int b = idx / (IDIM * SEQ);
    int r = idx % (IDIM * SEQ);
    int i = r / SEQ;
    int t = r % SEQ;
    xT[idx] = (__bf16)x[((size_t)b * SEQ + t) * IDIM + i];
}

// phiRep[pk][s][i] = phiRev_pk[i+s], phiRev_pk[li] = sgn*phi[PHID-li, kf]
__global__ void prep_phirep_kernel(const float* __restrict__ phi, __bf16* __restrict__ rep) {
    int idx = blockIdx.x * blockDim.x + threadIdx.x;     // over 2*NF*8*PHIN
    if (idx >= 2 * NF * 8 * PHIN) return;
    int pk = idx / (8 * PHIN);
    int r  = idx % (8 * PHIN);
    int s  = r / PHIN;
    int i  = r % PHIN;
    int p  = pk / NF;
    int kf = pk % NF;
    int li = i + s;
    float v = 0.0f;
    int l = PHID - li;
    if (l >= 0 && l < SEQ) {
        float ph = phi[(size_t)l * NF + kf];
        v = (p && (l & 1)) ? -ph : ph;
    }
    rep[idx] = (__bf16)v;
}

// M_plus/M_minus [NF,I,O] f32 -> McatT [O][KDIM] bf16  (kk = p*6144 + k*256 + i)
__global__ void prep_mT_kernel(const float* __restrict__ mp, const float* __restrict__ mm,
                               __bf16* __restrict__ mt) {
    int idx = blockIdx.x * blockDim.x + threadIdx.x;     // over ODIM*KDIM, kk fastest
    if (idx >= ODIM * KDIM) return;
    int o  = idx / KDIM;
    int kk = idx % KDIM;
    int p  = kk / (NF * IDIM);
    int j  = kk % (NF * IDIM);
    const float* src = p ? mm : mp;
    mt[idx] = (__bf16)src[(size_t)j * ODIM + o];
}

// ---------------------------------------------------------------------------
// Phase 1: transposed Toeplitz conv.
//   grid = (BATCH, SEQ/16, 2*NF), block = 256 (8 waves).
// ---------------------------------------------------------------------------
__global__ __launch_bounds__(256) void conv_kernel(const __bf16* __restrict__ xT,
                                                   const __bf16* __restrict__ phiRep,
                                                   __bf16* __restrict__ U) {
    __shared__ __bf16 phiLDS[8 * PHIN];                  // 18,432 B

    const int pk   = blockIdx.z;
    const int b    = blockIdx.x;
    const int l0   = blockIdx.y * 16;
    const int wave = threadIdx.x >> 5;
    const int lane = threadIdx.x & 31;
    const int hl   = lane >> 4;
    const int m    = lane & 15;

    // ---- stage the 18,432B filter table into LDS -------------------------
#if __has_builtin(__builtin_amdgcn_tensor_load_to_lds)
    if (threadIdx.x < 32) {                              // wave 0 issues the TDM op
        const unsigned long long ga =
            (unsigned long long)(const void*)(phiRep + (size_t)pk * 8 * PHIN);
        typedef __attribute__((address_space(3))) const __bf16* lds_cptr;
        const unsigned ldsAddr = (unsigned)(size_t)(lds_cptr)phiLDS;

        v4u g0;
        g0[0] = 1u;                                        // count=1 (valid user D#)
        g0[1] = ldsAddr;                                   // lds_addr
        g0[2] = (unsigned)(ga & 0xffffffffu);              // global_addr[31:0]
        g0[3] = (unsigned)((ga >> 32) & 0x01ffffffu)       // global_addr[56:32]
              | (2u << 30);                                // type=2 ("image")
        v8i g1;
        g1[0] = 0x30000;                                   // data_size=3 (8B elems)
        g1[1] = (int)(2304u << 16);                        // tensor_dim0[15:0]=2304
        g1[2] = (int)(1u << 16);                           // dim0 hi=0 | tensor_dim1=1
        g1[3] = (int)(2304u << 16);                        // dim1 hi=0 | tile_dim0=2304
        g1[4] = 1;                                         // tile_dim1=1, tile_dim2=0
        g1[5] = 2304;                                      // tensor_dim0_stride lo32
        g1[6] = 0;
        g1[7] = 0;
        v4i g2 = {0, 0, 0, 0};
        v4i g3 = {0, 0, 0, 0};
#if __clang_major__ >= 23
        v8i g4 = {0, 0, 0, 0, 0, 0, 0, 0};
        __builtin_amdgcn_tensor_load_to_lds(g0, g1, g2, g3, g4, 0);
#else
        __builtin_amdgcn_tensor_load_to_lds(g0, g1, g2, g3, 0);
#endif
        __builtin_amdgcn_s_wait_tensorcnt(0);
    }
#else
    {
        const uint4* src = (const uint4*)(phiRep + (size_t)pk * 8 * PHIN);
        uint4* dst = (uint4*)phiLDS;
        for (int i = threadIdx.x; i < (8 * PHIN * 2) / 16; i += 256)
            dst[i] = src[i];
    }
#endif
    __syncthreads();

    const int i0 = wave * 32;
    const __bf16* arow0 = xT + ((size_t)b * IDIM + i0 + m) * SEQ;   // A tile0 row
    const __bf16* arow1 = arow0 + (size_t)16 * SEQ;                 // A tile1 row

    // B base: logical reversed index at t0=0 for this lane, shift-aligned to 8
    const int base0 = PHID - l0 - m + hl * 16;
    const int sft   = base0 & 7;
    const __bf16* brow = phiLDS + (size_t)sft * PHIN + (base0 - sft);

    v8f acc0 = {}, acc1 = {};
    const int tEnd = l0 + 16;

#pragma unroll 2
    for (int t0 = 0; t0 < tEnd; t0 += 32) {
        union { uint4 q[2]; v16bf v; } a0, a1, bb;
        const uint4* pa0 = (const uint4*)(arow0 + t0 + hl * 16);
        const uint4* pa1 = (const uint4*)(arow1 + t0 + hl * 16);
        a0.q[0] = pa0[0]; a0.q[1] = pa0[1];
        a1.q[0] = pa1[0]; a1.q[1] = pa1[1];
        const uint4* pb = (const uint4*)(brow + t0);
        bb.q[0] = pb[0]; bb.q[1] = pb[1];
        acc0 = __builtin_amdgcn_wmma_f32_16x16x32_bf16(false, a0.v, false, bb.v,
                                                       (short)0, acc0, false, false);
        acc1 = __builtin_amdgcn_wmma_f32_16x16x32_bf16(false, a1.v, false, bb.v,
                                                       (short)0, acc1, false, false);
    }

    // C store: lane -> col l = l0+m; rows i consecutive -> pack 8 bf16 = b128
    union { __bf16 h[8]; uint4 q; } s0, s1;
#pragma unroll
    for (int v = 0; v < 8; ++v) {
        s0.h[v] = (__bf16)acc0[v];
        s1.h[v] = (__bf16)acc1[v];
    }
    __bf16* dst = U + ((size_t)b * SEQ + l0 + m) * KDIM
                    + (size_t)pk * IDIM + i0 + hl * 8;
    *(uint4*)dst        = s0.q;
    *(uint4*)(dst + 16) = s1.q;
}

// ---------------------------------------------------------------------------
// Phase 2: projection GEMM.  out[bl, o] = sum_kk U[bl, kk] * McatT[o, kk]
// M = 4096, N = 256, K = 12288.  Wave tile 32x64 (2 A x 4 B frags, 8 wmma),
// 8 waves as 4(M) x 2(N) -> block tile 128x128; in-kernel col-block loop so
// U (100MB) is streamed from HBM once.  grid = (32).
// ---------------------------------------------------------------------------
__global__ __launch_bounds__(256) void proj_kernel(const __bf16* __restrict__ U,
                                                   const __bf16* __restrict__ McT,
                                                   float* __restrict__ out) {
    const int wave = threadIdx.x >> 5;
    const int lane = threadIdx.x & 31;
    const int hl   = lane >> 4;
    const int m    = lane & 15;
    const int wm   = wave & 3;
    const int wn   = wave >> 2;

    const int row0 = blockIdx.x * 128 + wm * 32;         // bl rows
    const __bf16* ua0 = U + (size_t)(row0 + m) * KDIM;
    const __bf16* ua1 = ua0 + (size_t)16 * KDIM;

    for (int cb = 0; cb < 2; ++cb) {
        const int col0 = cb * 128 + wn * 64;             // o cols
        const __bf16* mb = McT + (size_t)(col0 + m) * KDIM + hl * 16;

        v8f acc[2][4] = {};

#pragma unroll 2
        for (int k0 = 0; k0 < KDIM; k0 += 32) {
            // prefetch the streamed U rows ahead (global_prefetch_b8)
            __builtin_prefetch(ua0 + k0 + 256, 0, 1);
            __builtin_prefetch(ua1 + k0 + 256, 0, 1);

            union { uint4 q[2]; v16bf v; } a0, a1;
            const uint4* pa0 = (const uint4*)(ua0 + k0 + hl * 16);
            const uint4* pa1 = (const uint4*)(ua1 + k0 + hl * 16);
            a0.q[0] = pa0[0]; a0.q[1] = pa0[1];
            a1.q[0] = pa1[0]; a1.q[1] = pa1[1];

#pragma unroll
            for (int q = 0; q < 4; ++q) {
                v16bf bq = *(const v16bf*)(mb + (size_t)(q * 16) * KDIM + k0);
                acc[0][q] = __builtin_amdgcn_wmma_f32_16x16x32_bf16(
                    false, a0.v, false, bq, (short)0, acc[0][q], false, false);
                acc[1][q] = __builtin_amdgcn_wmma_f32_16x16x32_bf16(
                    false, a1.v, false, bq, (short)0, acc[1][q], false, false);
            }
        }

#pragma unroll
        for (int q = 0; q < 4; ++q) {
            const int c = col0 + q * 16 + m;
#pragma unroll
            for (int v = 0; v < 8; ++v) {
                const int r = row0 + v + hl * 8;
                out[(size_t)r * ODIM + c]        = acc[0][q][v];
                out[(size_t)(r + 16) * ODIM + c] = acc[1][q][v];
            }
        }
    }
}

// ---------------------------------------------------------------------------
extern "C" void kernel_launch(void* const* d_in, const int* in_sizes, int n_in,
                              void* d_out, int out_size, void* d_ws, size_t ws_size,
                              hipStream_t stream) {
    const float* x   = (const float*)d_in[0];   // [4, 1024, 256]
    const float* phi = (const float*)d_in[1];   // [1024, 24]
    const float* mp  = (const float*)d_in[2];   // [24, 256, 256]
    const float* mm  = (const float*)d_in[3];   // [24, 256, 256]
    float* out       = (float*)d_out;           // [4, 1024, 256]

    char* ws = (char*)d_ws;
    __bf16* xT     = (__bf16*)(ws + OFF_XT);
    __bf16* phiRep = (__bf16*)(ws + OFF_PHIREP);
    __bf16* mcT    = (__bf16*)(ws + OFF_MT);
    __bf16* ucat   = (__bf16*)(ws + OFF_UCAT);

    {   // x transpose + bf16
        const int n = BATCH * IDIM * SEQ;
        prep_xT_kernel<<<(n + 255) / 256, 256, 0, stream>>>(x, xT);
    }
    {   // reversed, zero-padded, shift-replicated filters
        const int n = 2 * NF * 8 * PHIN;
        prep_phirep_kernel<<<(n + 255) / 256, 256, 0, stream>>>(phi, phiRep);
    }
    {   // M transpose + bf16
        const int n = ODIM * KDIM;
        prep_mT_kernel<<<(n + 255) / 256, 256, 0, stream>>>(mp, mm, mcT);
    }

    {   // phase 1: Toeplitz conv -> U (bf16)
        dim3 grid(BATCH, SEQ / 16, 2 * NF);
        conv_kernel<<<grid, 256, 0, stream>>>(xT, phiRep, ucat);
    }
    {   // phase 2: projection GEMM -> out (f32)
        dim3 grid((BATCH * SEQ) / 128);
        proj_kernel<<<grid, 256, 0, stream>>>(ucat, mcT, out);
    }
}
